// LayoutLMv2Attention_16028817949015
// MI455X (gfx1250) — matrix operand order repaired and checked by reading, not compiled, not run on hardware
//
#include <hip/hip_runtime.h>
#include <cstdint>
#include <cstddef>

#define BATCH 8
#define SEQ   1024
#define HIDN  1024
#define NH    16
#define HD    64

typedef __attribute__((ext_vector_type(4)))  float  v4f;
typedef __attribute__((ext_vector_type(8)))  float  v8f;
typedef __attribute__((ext_vector_type(4)))  __bf16 v4bf;
typedef __attribute__((ext_vector_type(8)))  __bf16 v8bf;
typedef __attribute__((ext_vector_type(16))) __bf16 v16bf;

// D = A(16x32 bf16) * B(32x16 bf16) + C(16x16 f32)
__device__ __forceinline__ v8f wmma_bf(v16bf a, v16bf b, v8f c) {
  return __builtin_amdgcn_wmma_f32_16x16x32_bf16(
      /*neg_a=*/false, a, /*neg_b=*/false, b,
      /*c_mod=*/(short)0, c, /*reuse_a=*/false, /*reuse_b=*/false);
}

__device__ __forceinline__ v16bf cat8(v8bf a, v8bf b) {
  return __builtin_shufflevector(a, b, 0,1,2,3,4,5,6,7,8,9,10,11,12,13,14,15);
}

// bf16x3 split: v ~= (float)h + (float)l with ~2^-16 relative error
__device__ __forceinline__ void split2(float v, __bf16& h, __bf16& l) {
  __bf16 hh = (__bf16)v;
  h = hh;
  l = (__bf16)(v - (float)hh);
}

// CDNA5 async global->LDS copy (ASYNCcnt-tracked), 16B per lane.
// VDST = per-lane LDS byte address, VADDR = per-lane 64-bit global address.
__device__ __forceinline__ void async_ld_b128(uint32_t lds_off, uint64_t gaddr) {
  asm volatile("global_load_async_to_lds_b128 %0, %1, off"
               :: "v"(lds_off), "v"(gaddr) : "memory");
}
__device__ __forceinline__ void wait_async0() {
  asm volatile("s_wait_asynccnt 0x0" ::: "memory");
}

// ---------------------------------------------------------------------------
// Kernel 0: split an fp32 tensor into hi/lo bf16 planes (vectorized x4).
// ---------------------------------------------------------------------------
__global__ void __launch_bounds__(256)
cvt_split4(const float* __restrict__ src, __bf16* __restrict__ hi,
           __bf16* __restrict__ lo, int n4)
{
  int i = blockIdx.x * 256 + threadIdx.x;
  if (i < n4) {
    v4f v = *reinterpret_cast<const v4f*>(src + 4 * (size_t)i);
    v4bf hh, ll;
    #pragma unroll
    for (int e = 0; e < 4; ++e) { __bf16 a, b; split2(v[e], a, b); hh[e] = a; ll[e] = b; }
    *reinterpret_cast<v4bf*>(hi + 4 * (size_t)i) = hh;
    *reinterpret_cast<v4bf*>(lo + 4 * (size_t)i) = ll;
  }
}

// ---------------------------------------------------------------------------
// Kernel 1: QKV projection, bf16x3.  One wave per 16x16 tile.
// Epilogue folds q_bias / v_bias / 0.125 scale and re-splits into bf16 hi/lo:
//   Q,K -> [B,NH,S,HD]   V -> transposed [B,NH,HD,S] (contiguous keys for PV)
// ---------------------------------------------------------------------------
__global__ void __launch_bounds__(32)
qkv_gemm(const __bf16* __restrict__ xh, const __bf16* __restrict__ xl,
         const __bf16* __restrict__ wh, const __bf16* __restrict__ wl,
         const float* __restrict__ qb, const float* __restrict__ vb,
         __bf16* __restrict__ Qh, __bf16* __restrict__ Ql,
         __bf16* __restrict__ Kh, __bf16* __restrict__ Kl,
         __bf16* __restrict__ Vth, __bf16* __restrict__ Vtl)
{
  const int lane = threadIdx.x & 31;
  const int col  = lane & 15;
  const int half = lane >> 4;
  const int mbase = blockIdx.x * 16;     // row in [0, B*S)
  const int obase = blockIdx.y * 16;     // col in [0, 3*HID)

  const __bf16* axh = xh + (size_t)(mbase + col) * HIDN;
  const __bf16* axl = xl + (size_t)(mbase + col) * HIDN;
  const __bf16* bwh = wh + (size_t)(obase + col) * HIDN;
  const __bf16* bwl = wl + (size_t)(obase + col) * HIDN;

  v8f acc = {};
  #pragma unroll 4
  for (int k = 0; k < HIDN; k += 32) {
    v16bf Ah = *reinterpret_cast<const v16bf*>(axh + k + 16 * half);
    v16bf Al = *reinterpret_cast<const v16bf*>(axl + k + 16 * half);
    v16bf Bh = *reinterpret_cast<const v16bf*>(bwh + k + 16 * half);
    v16bf Bl = *reinterpret_cast<const v16bf*>(bwl + k + 16 * half);
    acc = wmma_bf(Ah, Bh, acc);
    acc = wmma_bf(Ah, Bl, acc);
    acc = wmma_bf(Al, Bh, acc);
  }

  const int part = obase >> 10;                 // 0=Q, 1=K, 2=V
  const int o_in = (obase & 1023) + col;
  const int h    = (obase & 1023) >> 6;
  const int hd   = (obase & 63) + col;
  const int b    = mbase >> 10;
  const int s0   = (mbase & 1023) + 8 * half;

  if (part == 0) {
    const float bias = qb[o_in];
    __bf16* dh = Qh + ((size_t)(b * NH + h) * SEQ) * HD + hd;
    __bf16* dl = Ql + ((size_t)(b * NH + h) * SEQ) * HD + hd;
    #pragma unroll
    for (int r = 0; r < 8; ++r) {
      __bf16 hh, ll;
      split2((acc[r] + bias) * 0.125f, hh, ll);   // 1/sqrt(HD) folded in
      dh[(size_t)(s0 + r) * HD] = hh;
      dl[(size_t)(s0 + r) * HD] = ll;
    }
  } else if (part == 1) {
    __bf16* dh = Kh + ((size_t)(b * NH + h) * SEQ) * HD + hd;
    __bf16* dl = Kl + ((size_t)(b * NH + h) * SEQ) * HD + hd;
    #pragma unroll
    for (int r = 0; r < 8; ++r) {
      __bf16 hh, ll;
      split2(acc[r], hh, ll);
      dh[(size_t)(s0 + r) * HD] = hh;
      dl[(size_t)(s0 + r) * HD] = ll;
    }
  } else {
    const float bias = vb[o_in];
    v8bf hh8, ll8;
    #pragma unroll
    for (int r = 0; r < 8; ++r) {
      __bf16 hh, ll;
      split2(acc[r] + bias, hh, ll);
      hh8[r] = hh; ll8[r] = ll;
    }
    const size_t off = ((size_t)(b * NH + h) * HD + hd) * SEQ + s0;
    *reinterpret_cast<v8bf*>(Vth + off) = hh8;
    *reinterpret_cast<v8bf*>(Vtl + off) = ll8;
  }
}

// ---------------------------------------------------------------------------
// Kernel 2: fused flash attention, bf16x3, 32-key tiles.
// 4 waves / workgroup share one (b,h); each wave owns a 16-query tile.
// K/V hi+lo tiles are staged into LDS with global_load_async_to_lds_b128
// (double-buffered, s_wait_asynccnt + barrier) -> 4x reuse out of LDS.
// S^T = K * Q^T puts the key axis in the per-lane VGPR dim; softmax is
// per-lane + one shfl_xor(16). P^T (hi/lo in-register) feeds ctx^T = V^T*P^T.
// ---------------------------------------------------------------------------
#define KT 32                       // keys per tile
#define PLANE_ELEMS (KT * HD)       // 2048 bf16 = 4KB per plane

__global__ void __launch_bounds__(128)
flash_attn(const __bf16* __restrict__ Qh, const __bf16* __restrict__ Ql,
           const __bf16* __restrict__ Kh, const __bf16* __restrict__ Kl,
           const __bf16* __restrict__ Vth, const __bf16* __restrict__ Vtl,
           const float* __restrict__ relp, const float* __restrict__ rel2,
           const unsigned char* __restrict__ mask,
           __bf16* __restrict__ ctxh, __bf16* __restrict__ ctxl)
{
  __shared__ __bf16 smem[2][4][PLANE_ELEMS];   // [buf][Kh,Kl,Vh,Vl][...]

  const int lane = threadIdx.x & 31;
  const int wave = threadIdx.x >> 5;           // 0..3: also the plane it loads
  const int col  = lane & 15;
  const int half = lane >> 4;
  const int h = blockIdx.y, b = blockIdx.z;
  const int q0 = (blockIdx.x * 4 + wave) * 16; // this wave's query tile
  const bool lolane = (half == 0);

  const size_t hb = (size_t)(b * NH + h);
  const __bf16* Qhp = Qh + hb * SEQ * HD;
  const __bf16* Qlp = Ql + hb * SEQ * HD;
  const __bf16* Khp = Kh + hb * SEQ * HD;
  const __bf16* Klp = Kl + hb * SEQ * HD;
  const __bf16* Vhp = Vth + hb * HD * SEQ;
  const __bf16* Vlp = Vtl + hb * HD * SEQ;

  const uint32_t lds0 = (uint32_t)(uintptr_t)&smem[0][0][0];

  // Async-stage one 32-key tile: wave w loads plane w (4KB = 8 x 32lanes x 16B)
  auto stage_tile = [&](int k0, int buf) {
    const uint32_t pbase = lds0 + (uint32_t)buf * (4u * PLANE_ELEMS * 2u) +
                           (uint32_t)wave * (PLANE_ELEMS * 2u);
    if (wave < 2) {
      // K planes: [S,HD] rows k0..k0+31 are one contiguous 4KB block
      const __bf16* g = (wave == 0 ? Khp : Klp) + (size_t)k0 * HD;
      #pragma unroll
      for (int i = 0; i < 8; ++i) {
        const uint32_t eo = i * 256 + lane * 8;
        async_ld_b128(pbase + eo * 2, (uint64_t)(uintptr_t)(g + eo));
      }
    } else {
      // V planes: [HD,S] -> tile rows d=0..63, 32 keys (64B) each
      const __bf16* g = (wave == 2 ? Vhp : Vlp);
      #pragma unroll
      for (int i = 0; i < 8; ++i) {
        const uint32_t eo = i * 256 + lane * 8;
        const uint32_t d = eo >> 5, kk = eo & 31;
        async_ld_b128(pbase + eo * 2,
                      (uint64_t)(uintptr_t)(g + (size_t)d * SEQ + k0 + kk));
      }
    }
  };

  // Q^T B-operands (K-dim = head dim, 2 chunks of 32), loaded once
  v16bf bqh[2], bql[2];
  #pragma unroll
  for (int c = 0; c < 2; ++c) {
    const size_t qoff = (size_t)(q0 + col) * HD + 32 * c + 8 * half;
    bqh[c] = cat8(*reinterpret_cast<const v8bf*>(Qhp + qoff),
                  *reinterpret_cast<const v8bf*>(Qhp + qoff + 16));
    bql[c] = cat8(*reinterpret_cast<const v8bf*>(Qlp + qoff),
                  *reinterpret_cast<const v8bf*>(Qlp + qoff + 16));
  }

  v8f acc[4] = {v8f{}, v8f{}, v8f{}, v8f{}};   // ctx^T, 4 chunks of 16 dims
  float m_i = -3.0e38f, l_i = 0.0f;

  const float* rprow = relp + ((size_t)h * SEQ + q0 + col) * SEQ;
  const float* r2row = rel2 + ((size_t)h * SEQ + q0 + col) * SEQ;
  const unsigned char* mrow = mask + (size_t)b * SEQ;

  stage_tile(0, 0);
  wait_async0();
  __syncthreads();

  for (int kt = 0; kt < SEQ / KT; ++kt) {
    const int k0 = kt * KT;
    const int buf = kt & 1;
    if (kt + 1 < SEQ / KT) {
      stage_tile(k0 + KT, buf ^ 1);           // prefetch next tile into LDS
      __builtin_prefetch(rprow + k0 + KT, 0, 1);
      __builtin_prefetch(r2row + k0 + KT, 0, 1);
    }
    const __bf16* kh_t = &smem[buf][0][0];
    const __bf16* kl_t = &smem[buf][1][0];
    const __bf16* vh_t = &smem[buf][2][0];
    const __bf16* vl_t = &smem[buf][3][0];

    // ---- S^T: two 16-key subtiles x 16 queries (K/V read from LDS) ----
    v8f st0 = {}, st1 = {};
    #pragma unroll
    for (int c = 0; c < 2; ++c) {
      const __bf16* r0h = kh_t + (size_t)col * HD + 32 * c;
      const __bf16* r0l = kl_t + (size_t)col * HD + 32 * c;
      v16bf a0h = cat8(*reinterpret_cast<const v8bf*>(r0h + 8 * half),
                       *reinterpret_cast<const v8bf*>(r0h + 16 + 8 * half));
      v16bf a0l = cat8(*reinterpret_cast<const v8bf*>(r0l + 8 * half),
                       *reinterpret_cast<const v8bf*>(r0l + 16 + 8 * half));
      st0 = wmma_bf(a0h, bqh[c], st0);
      st0 = wmma_bf(a0h, bql[c], st0);
      st0 = wmma_bf(a0l, bqh[c], st0);
      const __bf16* r1h = kh_t + (size_t)(16 + col) * HD + 32 * c;
      const __bf16* r1l = kl_t + (size_t)(16 + col) * HD + 32 * c;
      v16bf a1h = cat8(*reinterpret_cast<const v8bf*>(r1h + 8 * half),
                       *reinterpret_cast<const v8bf*>(r1h + 16 + 8 * half));
      v16bf a1l = cat8(*reinterpret_cast<const v8bf*>(r1l + 8 * half),
                       *reinterpret_cast<const v8bf*>(r1l + 16 + 8 * half));
      st1 = wmma_bf(a1h, bqh[c], st1);
      st1 = wmma_bf(a1h, bql[c], st1);
      st1 = wmma_bf(a1l, bqh[c], st1);
    }

    // ---- rel_pos + rel_2d_pos + mask (lane holds keys k0+8*half+r, +16) ----
    v4f ra0 = *reinterpret_cast<const v4f*>(rprow + k0 + 8 * half);
    v4f ra1 = *reinterpret_cast<const v4f*>(rprow + k0 + 8 * half + 4);
    v4f ra2 = *reinterpret_cast<const v4f*>(rprow + k0 + 16 + 8 * half);
    v4f ra3 = *reinterpret_cast<const v4f*>(rprow + k0 + 16 + 8 * half + 4);
    v4f rb0 = *reinterpret_cast<const v4f*>(r2row + k0 + 8 * half);
    v4f rb1 = *reinterpret_cast<const v4f*>(r2row + k0 + 8 * half + 4);
    v4f rb2 = *reinterpret_cast<const v4f*>(r2row + k0 + 16 + 8 * half);
    v4f rb3 = *reinterpret_cast<const v4f*>(r2row + k0 + 16 + 8 * half + 4);
    uint64_t mk0 = *reinterpret_cast<const uint64_t*>(mrow + k0 + 8 * half);
    uint64_t mk1 = *reinterpret_cast<const uint64_t*>(mrow + k0 + 16 + 8 * half);

    float sv0[8], sv1[8];
    #pragma unroll
    for (int r = 0; r < 8; ++r) {
      float a0 = (r < 4) ? (ra0[r] + rb0[r]) : (ra1[r - 4] + rb1[r - 4]);
      float a1 = (r < 4) ? (ra2[r] + rb2[r]) : (ra3[r - 4] + rb3[r - 4]);
      sv0[r] = st0[r] + a0;
      sv1[r] = st1[r] + a1;
      if ((mk0 >> (8 * r)) & 0xFFull) sv0[r] = -3.0e38f;
      if ((mk1 >> (8 * r)) & 0xFFull) sv1[r] = -3.0e38f;
    }

    // ---- online softmax ----
    float lm = sv0[0];
    #pragma unroll
    for (int r = 1; r < 8; ++r) lm = fmaxf(lm, sv0[r]);
    #pragma unroll
    for (int r = 0; r < 8; ++r) lm = fmaxf(lm, sv1[r]);
    lm = fmaxf(lm, __shfl_xor(lm, 16, 32));
    const float mnew = fmaxf(m_i, lm);

    float p0[8], p1[8], rs = 0.0f;
    #pragma unroll
    for (int r = 0; r < 8; ++r) {
      p0[r] = __expf(sv0[r] - mnew);
      p1[r] = __expf(sv1[r] - mnew);
      rs += p0[r] + p1[r];
    }
    rs += __shfl_xor(rs, 16, 32);
    const float sc = __expf(m_i - mnew);
    l_i = l_i * sc + rs;
    m_i = mnew;

    #pragma unroll
    for (int dc = 0; dc < 4; ++dc)
      #pragma unroll
      for (int r = 0; r < 8; ++r) acc[dc][r] *= sc;

    // ---- P^T -> bf16 hi/lo B operand (keys 0..15 lanes<16, 16..31 lanes>=16)
    v16bf bph, bpl;
    #pragma unroll
    for (int e = 0; e < 8; ++e) {
      float oth = __shfl_xor(p1[e], 16, 32);
      float v = lolane ? p0[e] : oth;
      __bf16 hh, ll; split2(v, hh, ll);
      bph[e] = hh; bpl[e] = ll;
    }
    #pragma unroll
    for (int e = 0; e < 8; ++e) {
      float oth = __shfl_xor(p0[e], 16, 32);
      float v = lolane ? oth : p1[e];
      __bf16 hh, ll; split2(v, hh, ll);
      bph[8 + e] = hh; bpl[8 + e] = ll;
    }

    // ---- ctx^T += V^T * P^T  (V from LDS: [64 dims][32 keys]) ----
    #pragma unroll
    for (int dc = 0; dc < 4; ++dc) {
      const __bf16* vh = vh_t + (size_t)(16 * dc + col) * KT;
      const __bf16* vl = vl_t + (size_t)(16 * dc + col) * KT;
      v16bf ah = *reinterpret_cast<const v16bf*>(vh + 16 * half);
      v16bf al = *reinterpret_cast<const v16bf*>(vl + 16 * half);
      acc[dc] = wmma_bf(ah, bph, acc[dc]);
      acc[dc] = wmma_bf(ah, bpl, acc[dc]);
      acc[dc] = wmma_bf(al, bph, acc[dc]);
    }

    wait_async0();        // own async loads (next tile) have landed
    __syncthreads();      // all waves done reading buf & done loading buf^1
  }

  // ---- normalize, split hi/lo, store ctx as [B,S,NH*HD] bf16 planes ----
  const float inv = 1.0f / l_i;
  const size_t cbase = (size_t)(b * SEQ + q0 + col) * HIDN + h * HD + 8 * half;
  #pragma unroll
  for (int dc = 0; dc < 4; ++dc) {
    v4bf h0, l0, h1, l1;
    #pragma unroll
    for (int r = 0; r < 4; ++r) {
      __bf16 hh, ll;
      split2(acc[dc][r] * inv, hh, ll);
      h0[r] = hh; l0[r] = ll;
      split2(acc[dc][r + 4] * inv, hh, ll);
      h1[r] = hh; l1[r] = ll;
    }
    *reinterpret_cast<v4bf*>(ctxh + cbase + 16 * dc)     = h0;
    *reinterpret_cast<v4bf*>(ctxh + cbase + 16 * dc + 4) = h1;
    *reinterpret_cast<v4bf*>(ctxl + cbase + 16 * dc)     = l0;
    *reinterpret_cast<v4bf*>(ctxl + cbase + 16 * dc + 4) = l1;
  }
}

// ---------------------------------------------------------------------------
// Kernel 3: out = LN(ctx @ w_out^T + b_out + x), bf16x3 GEMM + fp32 LN.
// One workgroup (8 waves) owns a 16x1024 strip -> LN stats via ds_add_f32.
// ---------------------------------------------------------------------------
__global__ void __launch_bounds__(256)
out_proj_ln(const __bf16* __restrict__ ctxh, const __bf16* __restrict__ ctxl,
            const __bf16* __restrict__ woh, const __bf16* __restrict__ wol,
            const float* __restrict__ bo, const float* __restrict__ x,
            const float* __restrict__ gamma, const float* __restrict__ beta,
            float* __restrict__ out)
{
  __shared__ float s_sum[16];
  __shared__ float s_sq[16];
  const int lane = threadIdx.x & 31;
  const int wave = threadIdx.x >> 5;
  const int col  = lane & 15;
  const int half = lane >> 4;
  const int sbase = blockIdx.x * 16;

  if (threadIdx.x < 16) { s_sum[threadIdx.x] = 0.0f; s_sq[threadIdx.x] = 0.0f; }
  __syncthreads();

  v8f acc[8] = {v8f{}, v8f{}, v8f{}, v8f{}, v8f{}, v8f{}, v8f{}, v8f{}};
  const __bf16* ahr = ctxh + (size_t)(sbase + col) * HIDN;
  const __bf16* alr = ctxl + (size_t)(sbase + col) * HIDN;

  for (int k = 0; k < HIDN; k += 32) {
    v16bf Ah = *reinterpret_cast<const v16bf*>(ahr + k + 16 * half);
    v16bf Al = *reinterpret_cast<const v16bf*>(alr + k + 16 * half);
    #pragma unroll
    for (int t = 0; t < 8; ++t) {
      const int ob = wave * 128 + t * 16;
      v16bf Bh = *reinterpret_cast<const v16bf*>(woh + (size_t)(ob + col) * HIDN + k + 16 * half);
      v16bf Bl = *reinterpret_cast<const v16bf*>(wol + (size_t)(ob + col) * HIDN + k + 16 * half);
      acc[t] = wmma_bf(Ah, Bh, acc[t]);
      acc[t] = wmma_bf(Ah, Bl, acc[t]);
      acc[t] = wmma_bf(Al, Bh, acc[t]);
    }
  }

  // bias + residual (fp32)
  #pragma unroll
  for (int t = 0; t < 8; ++t) {
    const int o = wave * 128 + t * 16 + col;
    const float bias = bo[o];
    #pragma unroll
    for (int r = 0; r < 8; ++r) {
      const int s = sbase + r + 8 * half;
      acc[t][r] += bias + x[(size_t)s * HIDN + o];
    }
  }

  // LN partial sums per row
  #pragma unroll
  for (int r = 0; r < 8; ++r) {
    float ps = 0.0f, pq = 0.0f;
    #pragma unroll
    for (int t = 0; t < 8; ++t) { float v = acc[t][r]; ps += v; pq += v * v; }
    atomicAdd(&s_sum[r + 8 * half], ps);
    atomicAdd(&s_sq[r + 8 * half], pq);
  }
  __syncthreads();

  #pragma unroll
  for (int r = 0; r < 8; ++r) {
    const int row = r + 8 * half;
    const float mean = s_sum[row] * (1.0f / HIDN);
    const float var  = s_sq[row] * (1.0f / HIDN) - mean * mean;
    const float rstd = rsqrtf(var + 1e-12f);
    const int s = sbase + row;
    #pragma unroll
    for (int t = 0; t < 8; ++t) {
      const int o = wave * 128 + t * 16 + col;
      out[(size_t)s * HIDN + o] = (acc[t][r] - mean) * rstd * gamma[o] + beta[o];
    }
  }
}

// ---------------------------------------------------------------------------
extern "C" void kernel_launch(void* const* d_in, const int* in_sizes, int n_in,
                              void* d_out, int out_size, void* d_ws, size_t ws_size,
                              hipStream_t stream) {
  (void)in_sizes; (void)n_in; (void)out_size; (void)ws_size;
  const float*         x    = (const float*)d_in[0];
  const unsigned char* mask = (const unsigned char*)d_in[1];
  const float*         relp = (const float*)d_in[2];
  const float*         rel2 = (const float*)d_in[3];
  const float*         wqkv = (const float*)d_in[4];
  const float*         qb   = (const float*)d_in[5];
  const float*         vb   = (const float*)d_in[6];
  const float*         wout = (const float*)d_in[7];
  const float*         bo   = (const float*)d_in[8];
  const float*         g    = (const float*)d_in[9];
  const float*         be   = (const float*)d_in[10];
  float* out = (float*)d_out;

  __bf16* ws = (__bf16*)d_ws;
  const size_t E  = (size_t)BATCH * SEQ * HIDN;   // 8M elements
  const size_t M1 = (size_t)HIDN * HIDN;          // 1M elements
  __bf16 *Qh   = ws,          *Ql   = ws + E;
  __bf16 *Kh   = ws + 2 * E,  *Kl   = ws + 3 * E;
  __bf16 *Vth  = ws + 4 * E,  *Vtl  = ws + 5 * E;
  __bf16 *ctxh = ws + 6 * E,  *ctxl = ws + 7 * E;
  __bf16 *xh   = ws + 8 * E,  *xl   = ws + 9 * E;
  __bf16 *wqh  = ws + 10 * E, *wql  = ws + 10 * E + 3 * M1;
  __bf16 *woh  = ws + 10 * E + 6 * M1, *wol = ws + 10 * E + 7 * M1;

  cvt_split4<<<dim3((unsigned)((E / 4 + 255) / 256)), 256, 0, stream>>>(x, xh, xl, (int)(E / 4));
  cvt_split4<<<dim3((unsigned)((3 * M1 / 4 + 255) / 256)), 256, 0, stream>>>(wqkv, wqh, wql, (int)(3 * M1 / 4));
  cvt_split4<<<dim3((unsigned)((M1 / 4 + 255) / 256)), 256, 0, stream>>>(wout, woh, wol, (int)(M1 / 4));

  qkv_gemm<<<dim3(BATCH * SEQ / 16, 3 * HIDN / 16), 32, 0, stream>>>(
      xh, xl, wqh, wql, qb, vb, Qh, Ql, Kh, Kl, Vth, Vtl);
  flash_attn<<<dim3(SEQ / 16 / 4, NH, BATCH), 128, 0, stream>>>(
      Qh, Ql, Kh, Kl, Vth, Vtl, relp, rel2, mask, ctxh, ctxl);
  out_proj_ln<<<dim3(BATCH * SEQ / 16), 256, 0, stream>>>(
      ctxh, ctxl, woh, wol, bo, x, g, be, out);
}
